// RibonucleicAcidSE3Transformer_54116587929653
// MI455X (gfx1250) — matrix operand order repaired and checked by reading, not compiled, not run on hardware
//
#include <hip/hip_runtime.h>
#include <hip/hip_bf16.h>

// ---------------------------------------------------------------------------
// SE(3) graph transformer forward for MI455X (gfx1250, wave32).
//   - All six 64x64 projections per layer on v_wmma_f32_16x16x32_bf16
//     (bf16 inputs, f32 accumulate; residual fused via WMMA C operand).
//   - Weights staged to LDS via GLOBAL_LOAD_ASYNC_TO_LDS_B128 when the
//     toolchain declares the gfx1250 async builtins (guarded fallback).
//   - Edge phase: wave-per-edge, segment softmax with ordered-uint atomic
//     max + f32 atomic adds; radial MLP weights staged in LDS, gates
//     recomputed in the scatter pass (cheaper than 134MB of gate traffic).
// ---------------------------------------------------------------------------

typedef __attribute__((ext_vector_type(16))) __bf16 bf16x16;
typedef __attribute__((ext_vector_type(8)))  float  f32x8;

#if __has_builtin(__builtin_amdgcn_global_load_async_to_lds_b128) && \
    __has_builtin(__builtin_amdgcn_s_wait_asynccnt)
#define USE_ASYNC_LDS 1
typedef int v4i __attribute__((ext_vector_type(4)));
typedef __attribute__((address_space(1))) v4i as1_v4i;  // global (AS1)
typedef __attribute__((address_space(3))) v4i as3_v4i;  // LDS (AS3)
#endif

__device__ __forceinline__ __bf16 tobf(float f) { return (__bf16)f; }
__device__ __forceinline__ float fast_rcp(float x) { return __builtin_amdgcn_rcpf(x); }
__device__ __forceinline__ float sigmf(float x) { return fast_rcp(1.f + __expf(-x)); }
__device__ __forceinline__ float siluf(float x) { return x * sigmf(x); }

// monotone float <-> uint map so atomicMax(uint) implements float max.
// key(f) >= 0x00800000 for all finite f, so zero-init acts as -inf identity.
__device__ __forceinline__ unsigned fkey(float f) {
  unsigned b = __float_as_uint(f);
  return (b & 0x80000000u) ? ~b : (b | 0x80000000u);
}
__device__ __forceinline__ float fkeyinv(unsigned k) {
  unsigned b = (k & 0x80000000u) ? (k ^ 0x80000000u) : ~k;
  return __uint_as_float(b);
}

// Cooperative global->LDS stage (whole block). Uses gfx1250 async DMA path
// when available; ends with a workgroup barrier either way.
__device__ __forceinline__ void stage_lds(float* dst, const float* src, int nfloats) {
#if defined(USE_ASYNC_LDS)
  for (int i = threadIdx.x * 4; i < nfloats; i += blockDim.x * 4) {
    __builtin_amdgcn_global_load_async_to_lds_b128(
        (as1_v4i*)(src + i), (as3_v4i*)(dst + i), 0, 0);
  }
  __builtin_amdgcn_s_wait_asynccnt(0);
#else
  for (int i = threadIdx.x; i < nfloats; i += blockDim.x) dst[i] = src[i];
#endif
  __syncthreads();
}

// ---------------------------------------------------------------------------
// Small utility kernels
// ---------------------------------------------------------------------------
__global__ void fill_zero(float* __restrict__ p, int n) {
  int i = blockIdx.x * blockDim.x + threadIdx.x;
  if (i < n) p[i] = 0.f;
}

__global__ void coord_sum(const float* __restrict__ c, float* __restrict__ acc, int n) {
  __shared__ float sm[3][256];
  float sx = 0.f, sy = 0.f, sz = 0.f;
  for (int i = blockIdx.x * blockDim.x + threadIdx.x; i < n; i += gridDim.x * blockDim.x) {
    sx += c[i * 3 + 0]; sy += c[i * 3 + 1]; sz += c[i * 3 + 2];
  }
  sm[0][threadIdx.x] = sx; sm[1][threadIdx.x] = sy; sm[2][threadIdx.x] = sz;
  __syncthreads();
  for (int st = 128; st > 0; st >>= 1) {
    if ((int)threadIdx.x < st) {
      sm[0][threadIdx.x] += sm[0][threadIdx.x + st];
      sm[1][threadIdx.x] += sm[1][threadIdx.x + st];
      sm[2][threadIdx.x] += sm[2][threadIdx.x + st];
    }
    __syncthreads();
  }
  if (threadIdx.x == 0) {
    atomicAdd(&acc[0], sm[0][0]);
    atomicAdd(&acc[1], sm[1][0]);
    atomicAdd(&acc[2], sm[2][0]);
  }
}

__global__ void center_coords(const float* __restrict__ c, const float* __restrict__ acc,
                              float* __restrict__ x, int n3, float invn) {
  int i = blockIdx.x * blockDim.x + threadIdx.x;
  if (i >= n3) return;
  int k = i - (i / 3) * 3;
  x[i] = c[i] - acc[k] * invn;
}

__global__ void edge_geom(const float* __restrict__ x, const int* __restrict__ src,
                          const int* __restrict__ dst, float* __restrict__ dg,
                          float* __restrict__ rh, int E) {
  int e = blockIdx.x * blockDim.x + threadIdx.x;
  if (e >= E) return;
  int si = src[e], di = dst[e];
  float r0 = x[di * 3 + 0] - x[si * 3 + 0];
  float r1 = x[di * 3 + 1] - x[si * 3 + 1];
  float r2 = x[di * 3 + 2] - x[si * 3 + 2];
  float d = sqrtf(r0 * r0 + r1 * r1 + r2 * r2 + 1e-12f);
  float id = fast_rcp(d);
  dg[e] = d;
  rh[e * 3 + 0] = r0 * id;
  rh[e * 3 + 1] = r1 * id;
  rh[e * 3 + 2] = r2 * id;
}

// v[n,d,c] = x[n,d] * Win_v[c]   (layout (N,3,C))
__global__ void v_init(const float* __restrict__ x, const float* __restrict__ winv,
                       float* __restrict__ v, int n192) {
  int i = blockIdx.x * blockDim.x + threadIdx.x;
  if (i >= n192) return;
  v[i] = x[i >> 6] * winv[i & 63];
}

// v[n,:,c] *= sigmoid(|v[n,:,c]|^2)
__global__ void v_gate(float* __restrict__ v, int nC) {
  int i = blockIdx.x * blockDim.x + threadIdx.x;
  if (i >= nC) return;
  int n = i >> 6, c = i & 63;
  float* p = v + (size_t)n * 192 + c;
  float a = p[0], b = p[64], d = p[128];
  float g = sigmf(a * a + b * b + d * d);
  p[0] = a * g; p[64] = b * g; p[128] = d * g;
}

// out[n,d] = sum_c v[n,d,c] * Wout_v[c]
__global__ void out_proj(const float* __restrict__ v, const float* __restrict__ wout,
                         float* __restrict__ out, int n3) {
  int i = blockIdx.x * blockDim.x + threadIdx.x;
  if (i >= n3) return;
  const float4* p = (const float4*)(v + (size_t)i * 64);
  const float4* w = (const float4*)wout;
  float acc = 0.f;
#pragma unroll
  for (int q = 0; q < 16; ++q) {
    float4 a = p[q], b = w[q];
    acc += a.x * b.x + a.y * b.y + a.z * b.z + a.w * b.w;
  }
  out[i] = acc;
}

// ---------------------------------------------------------------------------
// WMMA GEMM: Out(nrows x 64) = act( A(nrows x 64) @ W(64 x 64) + Cbuf )
// W staged to LDS once per block (async DMA path when available); each wave
// owns one 16-col tile, builds both B fragments once, then grid-strides over
// row tiles: per tile only 8 A-loads (b128) + optional C-loads + 2 WMMA.
// Fragment k order: contiguous k = kb + 16*hi + i in both A and B (wave32).
// ---------------------------------------------------------------------------
__global__ void wmma_gemm64(const float* __restrict__ A, const float* __restrict__ W,
                            const float* Cbuf, float* Out, int nrows, int act) {
  __shared__ float sW[4096];
  stage_lds(sW, W, 4096);

  const int lane = threadIdx.x & 31;
  const int lo = lane & 15;
  const int hi = lane >> 4;
  const int wid = blockIdx.x * (blockDim.x >> 5) + (threadIdx.x >> 5);
  const int nw = gridDim.x * (blockDim.x >> 5);
  const int cb = (wid & 3) << 4;  // this wave's column tile

  // B 32x16 bf16 fragments (one per 32-wide K block), built once from LDS:
  // lane<16 -> K=kb+0..15 ; lane>=16 -> K=kb+16..31 ; N=lo
  bf16x16 bfr[2];
#pragma unroll
  for (int t = 0; t < 2; ++t) {
    const float* Wr = sW + (t * 32 + hi * 16) * 64 + cb + lo;
#pragma unroll
    for (int q = 0; q < 8; ++q) {
      bfr[t][2 * q + 0] = tobf(Wr[(2 * q + 0) * 64]);
      bfr[t][2 * q + 1] = tobf(Wr[(2 * q + 1) * 64]);
    }
  }

  const int nrt = nrows >> 4;
  for (int rt = wid >> 2; rt < nrt; rt += nw >> 2) {
    const int rb = rt << 4;
    f32x8 acc;
#pragma unroll
    for (int r = 0; r < 8; ++r)
      acc[r] = Cbuf ? Cbuf[(size_t)(rb + r + hi * 8) * 64 + cb + lo] : 0.f;

#pragma unroll
    for (int kb = 0; kb < 64; kb += 32) {
      // A 16x32 bf16 fragment: a[i]=A[row, kb+hi*16+i], i=0..15
      const float4* Ap0 = (const float4*)(A + (size_t)(rb + lo) * 64 + kb + hi * 16);
      const float4* Ap1 = (const float4*)(A + (size_t)(rb + lo) * 64 + kb + hi * 16 + 8);
      float4 f0 = Ap0[0], f1 = Ap0[1], f2 = Ap1[0], f3 = Ap1[1];
      bf16x16 a;
      a[0] = tobf(f0.x);  a[1] = tobf(f0.y);  a[2] = tobf(f0.z);  a[3] = tobf(f0.w);
      a[4] = tobf(f1.x);  a[5] = tobf(f1.y);  a[6] = tobf(f1.z);  a[7] = tobf(f1.w);
      a[8] = tobf(f2.x);  a[9] = tobf(f2.y);  a[10] = tobf(f2.z); a[11] = tobf(f2.w);
      a[12] = tobf(f3.x); a[13] = tobf(f3.y); a[14] = tobf(f3.z); a[15] = tobf(f3.w);
      acc = __builtin_amdgcn_wmma_f32_16x16x32_bf16(false, a, false, bfr[kb >> 5],
                                                    (short)0, acc, false, false);
    }

#pragma unroll
    for (int r = 0; r < 8; ++r) {
      float dv = acc[r];
      if (act == 1) dv = siluf(dv);
      Out[(size_t)(rb + r + hi * 8) * 64 + cb + lo] = dv;
    }
  }
}

// Same GEMM but A gathered on the fly:
//   A[row,k] = k<32 ? embed[atoms[row],k] : t_emb[t,k-32]
__global__ void wmma_gemm_sinit(const int* __restrict__ atoms,
                                const float* __restrict__ embed,
                                const float* __restrict__ temb,
                                const int* __restrict__ tptr,
                                const float* __restrict__ W,
                                float* __restrict__ Out, int nrows) {
  __shared__ float sW[4096];
  stage_lds(sW, W, 4096);

  const int lane = threadIdx.x & 31;
  const int lo = lane & 15;
  const int hi = lane >> 4;
  const int wid = blockIdx.x * (blockDim.x >> 5) + (threadIdx.x >> 5);
  const int nw = gridDim.x * (blockDim.x >> 5);
  const int cb = (wid & 3) << 4;
  const int tval = *tptr;
  const float* trow = temb + (size_t)tval * 32;

  bf16x16 bfr[2];
#pragma unroll
  for (int t = 0; t < 2; ++t) {
    const float* Wr = sW + (t * 32 + hi * 16) * 64 + cb + lo;
#pragma unroll
    for (int q = 0; q < 8; ++q) {
      bfr[t][2 * q + 0] = tobf(Wr[(2 * q + 0) * 64]);
      bfr[t][2 * q + 1] = tobf(Wr[(2 * q + 1) * 64]);
    }
  }

  const int nrt = nrows >> 4;
  for (int rt = wid >> 2; rt < nrt; rt += nw >> 2) {
    const int rb = rt << 4;
    const int row = rb + lo;
    const int at = atoms[row];
    const float* erow = embed + (size_t)at * 32;

    f32x8 acc;
#pragma unroll
    for (int r = 0; r < 8; ++r) acc[r] = 0.f;

#pragma unroll
    for (int kb = 0; kb < 64; kb += 32) {
      const float* Ar = (kb == 0) ? erow : trow;
      const float4* Ap0 = (const float4*)(Ar + hi * 16);
      const float4* Ap1 = (const float4*)(Ar + hi * 16 + 8);
      float4 f0 = Ap0[0], f1 = Ap0[1], f2 = Ap1[0], f3 = Ap1[1];
      bf16x16 a;
      a[0] = tobf(f0.x);  a[1] = tobf(f0.y);  a[2] = tobf(f0.z);  a[3] = tobf(f0.w);
      a[4] = tobf(f1.x);  a[5] = tobf(f1.y);  a[6] = tobf(f1.z);  a[7] = tobf(f1.w);
      a[8] = tobf(f2.x);  a[9] = tobf(f2.y);  a[10] = tobf(f2.z); a[11] = tobf(f2.w);
      a[12] = tobf(f3.x); a[13] = tobf(f3.y); a[14] = tobf(f3.z); a[15] = tobf(f3.w);
      acc = __builtin_amdgcn_wmma_f32_16x16x32_bf16(false, a, false, bfr[kb >> 5],
                                                    (short)0, acc, false, false);
    }
#pragma unroll
    for (int r = 0; r < 8; ++r)
      Out[(size_t)(rb + r + hi * 8) * 64 + cb + lo] = acc[r];
  }
}

// ---------------------------------------------------------------------------
// Edge pass A: attention logits + segment max. One wave per edge (grid-stride),
// 2 channels/lane; radial weights staged in LDS.
// ---------------------------------------------------------------------------
__global__ void edge_logits(const int* __restrict__ src, const int* __restrict__ dst,
                            const float* __restrict__ dg, const float* __restrict__ rh,
                            const float* __restrict__ qs, const float* __restrict__ ks,
                            const float* __restrict__ qv, const float* __restrict__ kv,
                            const float* __restrict__ RW1, const float* __restrict__ Rb1,
                            const float* __restrict__ RW2, const float* __restrict__ Rb2,
                            float* __restrict__ lg, unsigned* __restrict__ mkey, int E) {
  __shared__ float sRW2[2048];
  __shared__ float sR1[32], sB1[32], sB2[64];
  stage_lds(sRW2, RW2, 2048);
  if (threadIdx.x < 32) { sR1[threadIdx.x] = RW1[threadIdx.x]; sB1[threadIdx.x] = Rb1[threadIdx.x]; }
  if (threadIdx.x < 64) sB2[threadIdx.x] = Rb2[threadIdx.x];
  __syncthreads();

  const int lane = threadIdx.x & 31;
  const int wid = blockIdx.x * (blockDim.x >> 5) + (threadIdx.x >> 5);
  const int nw = gridDim.x * (blockDim.x >> 5);
  const int gb = lane & ~7;
  const float r1w = sR1[lane], b1w = sB1[lane];
  const float b2a = sB2[lane], b2b = sB2[lane + 32];

  for (int e = wid; e < E; e += nw) {
    const int si = src[e], di = dst[e];
    const float dd = dg[e];
    const float r0 = rh[e * 3 + 0], r1 = rh[e * 3 + 1], r2 = rh[e * 3 + 2];

    float hid = fmaxf(fmaf(dd, r1w, b1w), 0.f);
    float rada = b2a, radb = b2b;
    for (int j = 0; j < 32; ++j) {
      float hj = __shfl(hid, j, 32);
      rada = fmaf(hj, sRW2[j * 64 + lane], rada);
      radb = fmaf(hj, sRW2[j * 64 + lane + 32], radb);
    }

    const int c0 = lane, c1 = lane + 32;
    const float* ksp = ks + (size_t)si * 64;
    const float* kvp = kv + (size_t)si * 192;
    const float* qsp = qs + (size_t)di * 64;
    const float* qvp = qv + (size_t)di * 192;
    float ks0 = ksp[c0], ks1 = ksp[c1];
    float kv00 = kvp[c0], kv10 = kvp[64 + c0], kv20 = kvp[128 + c0];
    float kv01 = kvp[c1], kv11 = kvp[64 + c1], kv21 = kvp[128 + c1];
    float vd0 = kv00 * r0 + kv10 * r1 + kv20 * r2;
    float vd1 = kv01 * r0 + kv11 * r1 + kv21 * r2;

    float g0a = __shfl(rada, gb + 0, 32), g1a = __shfl(rada, gb + 1, 32);
    float g2a = __shfl(rada, gb + 2, 32), g3a = __shfl(rada, gb + 3, 32);
    float g0b = __shfl(radb, gb + 0, 32), g1b = __shfl(radb, gb + 1, 32);
    float g2b = __shfl(radb, gb + 2, 32), g3b = __shfl(radb, gb + 3, 32);

    float qs0 = qsp[c0], qs1 = qsp[c1];
    float qv00 = qvp[c0], qv10 = qvp[64 + c0], qv20 = qvp[128 + c0];
    float qv01 = qvp[c1], qv11 = qvp[64 + c1], qv21 = qvp[128 + c1];

    float lgA = qs0 * (g0a * ks0 + g1a * vd0)
              + qv00 * (g2a * kv00 + g3a * ks0 * r0)
              + qv10 * (g2a * kv10 + g3a * ks0 * r1)
              + qv20 * (g2a * kv20 + g3a * ks0 * r2);
    float lgB = qs1 * (g0b * ks1 + g1b * vd1)
              + qv01 * (g2b * kv01 + g3b * ks1 * r0)
              + qv11 * (g2b * kv11 + g3b * ks1 * r1)
              + qv21 * (g2b * kv21 + g3b * ks1 * r2);

    for (int o = 1; o < 8; o <<= 1) {
      lgA += __shfl_xor(lgA, o, 32);
      lgB += __shfl_xor(lgB, o, 32);
    }
    if ((lane & 7) == 0) {
      const int h = lane >> 3;
      const float sc = 0.35355339059327373f;  // 1/sqrt(Ch)
      float la = lgA * sc, lb = lgB * sc;
      lg[(size_t)e * 8 + h] = la;
      lg[(size_t)e * 8 + h + 4] = lb;
      atomicMax(&mkey[di * 8 + h], fkey(la));
      atomicMax(&mkey[di * 8 + h + 4], fkey(lb));
    }
  }
}

// Edge pass B: a = exp(logit - m[dst]); denom += a  (one thread per edge*head)
__global__ void edge_softmax(const int* __restrict__ dst, const unsigned* __restrict__ mkey,
                             float* __restrict__ lg, float* __restrict__ denom, int EH) {
  int i = blockIdx.x * blockDim.x + threadIdx.x;
  if (i >= EH) return;
  int e = i >> 3, h = i & 7;
  int di = dst[e];
  float mm = fkeyinv(mkey[di * 8 + h]);
  float a = __expf(lg[i] - mm);
  lg[i] = a;
  atomicAdd(&denom[di * 8 + h], a);
}

// Edge pass C: alpha-weighted scatter-add of val_s / val_v (radial gates 4..7
// recomputed from LDS-staged weights).
__global__ void edge_aggregate(const int* __restrict__ src, const int* __restrict__ dst,
                               const float* __restrict__ dg, const float* __restrict__ rh,
                               const float* __restrict__ ks, const float* __restrict__ kv,
                               const float* __restrict__ RW1, const float* __restrict__ Rb1,
                               const float* __restrict__ RW2, const float* __restrict__ Rb2,
                               const float* __restrict__ aexp, const float* __restrict__ denom,
                               float* __restrict__ aggs, float* __restrict__ aggv, int E) {
  __shared__ float sRW2[2048];
  __shared__ float sR1[32], sB1[32], sB2[64];
  stage_lds(sRW2, RW2, 2048);
  if (threadIdx.x < 32) { sR1[threadIdx.x] = RW1[threadIdx.x]; sB1[threadIdx.x] = Rb1[threadIdx.x]; }
  if (threadIdx.x < 64) sB2[threadIdx.x] = Rb2[threadIdx.x];
  __syncthreads();

  const int lane = threadIdx.x & 31;
  const int wid = blockIdx.x * (blockDim.x >> 5) + (threadIdx.x >> 5);
  const int nw = gridDim.x * (blockDim.x >> 5);
  const int gb = lane & ~7;
  const int h0 = lane >> 3;
  const float r1w = sR1[lane], b1w = sB1[lane];
  const float b2a = sB2[lane], b2b = sB2[lane + 32];

  for (int e = wid; e < E; e += nw) {
    const int si = src[e], di = dst[e];
    const float dd = dg[e];
    const float r0 = rh[e * 3 + 0], r1 = rh[e * 3 + 1], r2 = rh[e * 3 + 2];

    float hid = fmaxf(fmaf(dd, r1w, b1w), 0.f);
    float rada = b2a, radb = b2b;
    for (int j = 0; j < 32; ++j) {
      float hj = __shfl(hid, j, 32);
      rada = fmaf(hj, sRW2[j * 64 + lane], rada);
      radb = fmaf(hj, sRW2[j * 64 + lane + 32], radb);
    }

    const int c0 = lane, c1 = lane + 32;
    const float* ksp = ks + (size_t)si * 64;
    const float* kvp = kv + (size_t)si * 192;
    float ks0 = ksp[c0], ks1 = ksp[c1];
    float kv00 = kvp[c0], kv10 = kvp[64 + c0], kv20 = kvp[128 + c0];
    float kv01 = kvp[c1], kv11 = kvp[64 + c1], kv21 = kvp[128 + c1];
    float vd0 = kv00 * r0 + kv10 * r1 + kv20 * r2;
    float vd1 = kv01 * r0 + kv11 * r1 + kv21 * r2;

    float g4a = __shfl(rada, gb + 4, 32), g5a = __shfl(rada, gb + 5, 32);
    float g6a = __shfl(rada, gb + 6, 32), g7a = __shfl(rada, gb + 7, 32);
    float g4b = __shfl(radb, gb + 4, 32), g5b = __shfl(radb, gb + 5, 32);
    float g6b = __shfl(radb, gb + 6, 32), g7b = __shfl(radb, gb + 7, 32);

    float alA = aexp[(size_t)e * 8 + h0] * fast_rcp(denom[di * 8 + h0] + 1e-9f);
    float alB = aexp[(size_t)e * 8 + h0 + 4] * fast_rcp(denom[di * 8 + h0 + 4] + 1e-9f);

    atomicAdd(&aggs[(size_t)di * 64 + c0], alA * (g4a * ks0 + g5a * vd0));
    atomicAdd(&aggs[(size_t)di * 64 + c1], alB * (g4b * ks1 + g5b * vd1));

    float* av = aggv + (size_t)di * 192;
    atomicAdd(&av[c0],       alA * (g6a * kv00 + g7a * ks0 * r0));
    atomicAdd(&av[64 + c0],  alA * (g6a * kv10 + g7a * ks0 * r1));
    atomicAdd(&av[128 + c0], alA * (g6a * kv20 + g7a * ks0 * r2));
    atomicAdd(&av[c1],       alB * (g6b * kv01 + g7b * ks1 * r0));
    atomicAdd(&av[64 + c1],  alB * (g6b * kv11 + g7b * ks1 * r1));
    atomicAdd(&av[128 + c1], alB * (g6b * kv21 + g7b * ks1 * r2));
  }
}

// ---------------------------------------------------------------------------
extern "C" void kernel_launch(void* const* d_in, const int* in_sizes, int n_in,
                              void* d_out, int out_size, void* d_ws, size_t ws_size,
                              hipStream_t stream) {
  const float* coords = (const float*)d_in[0];
  const float* embed  = (const float*)d_in[1];
  const float* temb   = (const float*)d_in[2];
  const float* Win_s  = (const float*)d_in[3];
  const float* Win_v  = (const float*)d_in[4];
  const float* Wq_s   = (const float*)d_in[5];
  const float* Wq_v   = (const float*)d_in[6];
  const float* Wk_s   = (const float*)d_in[7];
  const float* Wk_v   = (const float*)d_in[8];
  const float* R_W1   = (const float*)d_in[9];
  const float* R_b1   = (const float*)d_in[10];
  const float* R_W2   = (const float*)d_in[11];
  const float* R_b2   = (const float*)d_in[12];
  const float* Wo_s   = (const float*)d_in[13];
  const float* Wo_v   = (const float*)d_in[14];
  const float* Wout_v = (const float*)d_in[15];
  const int* atoms = (const int*)d_in[16];
  const int* srcI  = (const int*)d_in[17];
  const int* dstI  = (const int*)d_in[18];
  const int* tptr  = (const int*)d_in[19];
  (void)n_in; (void)out_size; (void)ws_size;

  const int n = in_sizes[0] / 3;   // 16384
  const int E = in_sizes[17];      // 262144

  float* ws = (float*)d_ws;
  size_t off = 0;
  auto alloc = [&](size_t cnt) { float* p = ws + off; off += cnt; return p; };
  float* x    = alloc((size_t)n * 3);
  float* dg   = alloc((size_t)E);
  float* rh   = alloc((size_t)E * 3);
  float* s    = alloc((size_t)n * 64);
  float* v    = alloc((size_t)n * 192);
  float* qs   = alloc((size_t)n * 64);
  float* ks   = alloc((size_t)n * 64);
  float* qv   = alloc((size_t)n * 192);
  float* kv   = alloc((size_t)n * 192);
  float* lg   = alloc((size_t)E * 8);
  float* zb   = alloc((size_t)n * 272);  // [mkey(8) | denom(8) | aggs(64) | aggv(192)]
  unsigned* mkey = (unsigned*)zb;
  float* denom = zb + (size_t)n * 8;
  float* aggs  = zb + (size_t)n * 16;
  float* aggv  = zb + (size_t)n * 80;
  float* macc  = alloc(4);

  const int T = 256;
  const int gemmB = 512;   // grid-stride over row tiles; 8 waves/block
  const int edgeB = 1024;  // grid-stride wave-per-edge

  // preamble: center coords, edge geometry, s/v init
  fill_zero<<<1, 32, 0, stream>>>(macc, 4);
  coord_sum<<<64, T, 0, stream>>>(coords, macc, n);
  center_coords<<<(n * 3 + T - 1) / T, T, 0, stream>>>(coords, macc, x, n * 3, 1.0f / n);
  edge_geom<<<(E + T - 1) / T, T, 0, stream>>>(x, srcI, dstI, dg, rh, E);
  wmma_gemm_sinit<<<gemmB, T, 0, stream>>>(atoms, embed, temb, tptr, Win_s, s, n);
  v_init<<<(n * 192 + T - 1) / T, T, 0, stream>>>(x, Win_v, v, n * 192);

  for (int l = 0; l < 4; ++l) {
    const float* wq_s = Wq_s + (size_t)l * 4096;
    const float* wk_s = Wk_s + (size_t)l * 4096;
    const float* wq_v = Wq_v + (size_t)l * 4096;
    const float* wk_v = Wk_v + (size_t)l * 4096;
    const float* wo_s = Wo_s + (size_t)l * 4096;
    const float* wo_v = Wo_v + (size_t)l * 4096;
    const float* rw1 = R_W1 + (size_t)l * 32;
    const float* rb1 = R_b1 + (size_t)l * 32;
    const float* rw2 = R_W2 + (size_t)l * 2048;
    const float* rb2 = R_b2 + (size_t)l * 64;

    wmma_gemm64<<<gemmB, T, 0, stream>>>(s, wq_s, nullptr, qs, n, 0);
    wmma_gemm64<<<gemmB, T, 0, stream>>>(s, wk_s, nullptr, ks, n, 0);
    wmma_gemm64<<<gemmB, T, 0, stream>>>(v, wq_v, nullptr, qv, 3 * n, 0);
    wmma_gemm64<<<gemmB, T, 0, stream>>>(v, wk_v, nullptr, kv, 3 * n, 0);

    fill_zero<<<(n * 272 + T - 1) / T, T, 0, stream>>>(zb, n * 272);

    edge_logits<<<edgeB, T, 0, stream>>>(srcI, dstI, dg, rh, qs, ks, qv, kv,
                                         rw1, rb1, rw2, rb2, lg, mkey, E);
    edge_softmax<<<(E * 8 + T - 1) / T, T, 0, stream>>>(dstI, mkey, lg, denom, E * 8);
    edge_aggregate<<<edgeB, T, 0, stream>>>(srcI, dstI, dg, rh, ks, kv,
                                            rw1, rb1, rw2, rb2, lg, denom, aggs, aggv, E);

    // s = silu(s + aggs @ Wo_s)   (residual fused via WMMA C accumulator)
    wmma_gemm64<<<gemmB, T, 0, stream>>>(aggs, wo_s, s, s, n, 1);
    // vr = v + aggv @ Wo_v ; then gate
    wmma_gemm64<<<gemmB, T, 0, stream>>>(aggv, wo_v, v, v, 3 * n, 0);
    v_gate<<<(n * 64 + T - 1) / T, T, 0, stream>>>(v, n * 64);
  }

  out_proj<<<(n * 3 + T - 1) / T, T, 0, stream>>>(v, Wout_v, (float*)d_out, n * 3);
}
